// SolubilityGNN_74156905333221
// MI455X (gfx1250) — matrix operand order repaired and checked by reading, not compiled, not run on hardware
//
#include <hip/hip_runtime.h>

// ---------------------------------------------------------------------------
// SolubilityGNN for MI455X (gfx1250, wave32, WMMA).
// Strategy: all big GEMMs via v_wmma_f32_16x16x32_f16 (f16 inputs, f32 acc).
// BatchNorm needs global column stats -> two-pass; we RECOMPUTE the first GEMM
// in pass B instead of materializing the 435MB intermediate (compute is far
// cheaper than HBM on MI455X: 23.3 TB/s vs PFLOP-class WMMA).
// Workspace required: ~54 MB.
// ---------------------------------------------------------------------------

typedef __attribute__((ext_vector_type(16))) _Float16 v16h;
typedef __attribute__((ext_vector_type(8)))  float    v8f;
typedef __attribute__((ext_vector_type(4)))  float    v4f;

#define NN  100000               // nodes
#define NE  1600000              // edges (before self loops)
#define NR  (NE + NN)            // message rows (with self loops) = 1,700,000
#define NG  2048                 // graphs
#define FIN 160
#define DD  64
#define EFD 12

#define WMMA_F16(acc, a, b) \
  (acc) = __builtin_amdgcn_wmma_f32_16x16x32_f16(false, (a), false, (b), (short)0, (acc), false, false)

// ---------------- fragment helpers ----------------

// 8 consecutive f32 -> 8 halves (16B-aligned source)
__device__ __forceinline__ void run8_full(const float* __restrict__ p, _Float16* d) {
  v4f a = *(const v4f*)p;
  v4f b = *(const v4f*)(p + 4);
#pragma unroll
  for (int i = 0; i < 4; ++i) { d[i] = (_Float16)a[i]; d[4 + i] = (_Float16)b[i]; }
}
__device__ __forceinline__ void run8_guard(const float* __restrict__ p, int valid, _Float16* d) {
#pragma unroll
  for (int i = 0; i < 8; ++i) d[i] = (i < valid) ? (_Float16)p[i] : (_Float16)0.f;
}

// B fragment (32x16 K x N, f16) from LDS weights stored [64 cols][KP k-contig].
// Lanes 0-15: col=lane, K = kc*32 + 0..7,16..23 ; lanes 16-31: col=lane-16, K += 8.
__device__ __forceinline__ v16h bfrag(const _Float16* __restrict__ Ws, int KP, int kc, int nc, int lane) {
  const _Float16* p = Ws + ((nc << 4) + (lane & 15)) * KP + (kc << 5) + ((lane & 16) ? 8 : 0);
  v16h b;
#pragma unroll
  for (int i = 0; i < 8; ++i) { b[i] = p[i]; b[8 + i] = p[16 + i]; }
  return b;
}

// A layout (16x32, f16): lanes 0-15 row=lane, halves 0..7 -> K=base+0..7,
// halves 8..15 -> K=base+16..23 with base=0; lanes 16-31 same rows, base=8.

// ---------------- GEMM1 cores (recomputed in both passes) ----------------

// Edge message input row r: [ h[src[r]][0..63] | edge_attr[r][0..11] | 0 pad ], K padded to 96
__device__ __forceinline__ void edge_gemm1(const float* __restrict__ h, const float* __restrict__ ea,
                                           const int* __restrict__ srcIdx,
                                           const _Float16* __restrict__ W1s,
                                           int tile, int lane, v8f acc[4]) {
  int mr = lane & 15;
  int r = tile * 16 + mr;
  bool hasE = (r < NE);
  int src = hasE ? srcIdx[r] : (r - NE);
  const float* hrow = h + (size_t)src * DD;
  const float* earow = ea + (size_t)(hasE ? r : 0) * EFD;
  int base = (lane & 16) ? 8 : 0;
#pragma unroll
  for (int kc = 0; kc < 3; ++kc) {
    _Float16 t[16];
#pragma unroll
    for (int half = 0; half < 2; ++half) {
      int k = kc * 32 + base + half * 16;   // multiple of 8, runs never straddle 64
      _Float16* d = t + half * 8;
      if (k + 8 <= DD) {
        run8_full(hrow + k, d);
      } else {
        int off = k - DD;
        int valid = hasE ? (EFD - off) : 0;
        valid = valid < 0 ? 0 : (valid > 8 ? 8 : valid);
        run8_guard(earow + off, valid, d);
      }
    }
    v16h a;
#pragma unroll
    for (int i = 0; i < 16; ++i) a[i] = t[i];
#pragma unroll
    for (int nc = 0; nc < 4; ++nc) {
      v16h b = bfrag(W1s, 96, kc, nc, lane);
      WMMA_F16(acc[nc], a, b);
    }
  }
}

// Node update input row r: [ h[r][0..63] | aggr[r][0..63] ], K = 128
__device__ __forceinline__ void node_gemm1(const float* __restrict__ h, const float* __restrict__ aggr,
                                           const _Float16* __restrict__ W1s,
                                           int tile, int lane, v8f acc[4]) {
  int r = tile * 16 + (lane & 15);
  const float* hrow = h + (size_t)r * DD;
  const float* arow = aggr + (size_t)r * DD;
  int base = (lane & 16) ? 8 : 0;
#pragma unroll
  for (int kc = 0; kc < 4; ++kc) {
    _Float16 t[16];
#pragma unroll
    for (int half = 0; half < 2; ++half) {
      int k = kc * 32 + base + half * 16;
      const float* p = (k < DD) ? (hrow + k) : (arow + (k - DD));
      run8_full(p, t + half * 8);
    }
    v16h a;
#pragma unroll
    for (int i = 0; i < 16; ++i) a[i] = t[i];
#pragma unroll
    for (int nc = 0; nc < 4; ++nc) {
      v16h b = bfrag(W1s, 128, kc, nc, lane);
      WMMA_F16(acc[nc], a, b);
    }
  }
}

// ---------------- kernels ----------------

__global__ void __launch_bounds__(256) k_zero(float* __restrict__ p, long n) {
  long i = (long)blockIdx.x * blockDim.x + threadIdx.x;
  long s = (long)gridDim.x * blockDim.x;
  for (; i < n; i += s) p[i] = 0.f;
}

// h = x @ emb_w + emb_b   [NN,160]@[160,64]
__global__ void __launch_bounds__(256) k_embed(const float* __restrict__ x,
                                               const float* __restrict__ w,
                                               const float* __restrict__ bias,
                                               float* __restrict__ h) {
  __shared__ _Float16 Ws[DD * FIN];
  __shared__ float bs[DD];
  int tid = threadIdx.x;
  for (int idx = tid; idx < DD * FIN; idx += 256) {
    int col = idx / FIN, k = idx - col * FIN;
    Ws[idx] = (_Float16)w[k * DD + col];
  }
  if (tid < DD) bs[tid] = bias[tid];
  __syncthreads();
  int lane = tid & 31;
  int tile = blockIdx.x * 8 + (tid >> 5);
  if (tile < NN / 16) {
    int row0 = tile * 16;
    int base = (lane & 16) ? 8 : 0;
    const float* arow = x + (size_t)(row0 + (lane & 15)) * FIN;
    v8f acc[4] = {};
#pragma unroll
    for (int kc = 0; kc < 5; ++kc) {
      _Float16 t[16];
      run8_full(arow + kc * 32 + base, t);
      run8_full(arow + kc * 32 + base + 16, t + 8);
      v16h a;
#pragma unroll
      for (int i = 0; i < 16; ++i) a[i] = t[i];
#pragma unroll
      for (int nc = 0; nc < 4; ++nc) {
        v16h b = bfrag(Ws, FIN, kc, nc, lane);
        WMMA_F16(acc[nc], a, b);
      }
    }
    int cn = lane & 15;
#pragma unroll
    for (int nc = 0; nc < 4; ++nc) {
      int col = nc * 16 + cn;
      float bi = bs[col];
#pragma unroll
      for (int i = 0; i < 8; ++i)
        h[(size_t)(row0 + base + i) * DD + col] = acc[nc][i] + bi;
    }
  }
}

// pass A: compute t1 = ef@W1+b1 in registers, accumulate column sum/sumsq for BN
__global__ void __launch_bounds__(256) k_edge_passA(const float* __restrict__ h,
                                                    const float* __restrict__ ea,
                                                    const int* __restrict__ srcIdx,
                                                    const float* __restrict__ w1,
                                                    const float* __restrict__ b1,
                                                    float* __restrict__ gsum,
                                                    float* __restrict__ gsq) {
  __shared__ _Float16 W1s[DD * 96];
  __shared__ float bs[DD], ssum[DD], ssq[DD];
  int tid = threadIdx.x;
  for (int idx = tid; idx < DD * 96; idx += 256) {
    int col = idx / 96, k = idx - col * 96;
    W1s[idx] = (k < DD + EFD) ? (_Float16)w1[k * DD + col] : (_Float16)0.f;
  }
  if (tid < DD) { bs[tid] = b1[tid]; ssum[tid] = 0.f; ssq[tid] = 0.f; }
  __syncthreads();
  int lane = tid & 31;
  int tile = blockIdx.x * 8 + (tid >> 5);
  if (tile < NR / 16) {
    v8f acc[4] = {};
    edge_gemm1(h, ea, srcIdx, W1s, tile, lane, acc);
    int cn = lane & 15;
#pragma unroll
    for (int nc = 0; nc < 4; ++nc) {
      int col = nc * 16 + cn;
      float bi = bs[col];
      float s = 0.f, q = 0.f;
#pragma unroll
      for (int i = 0; i < 8; ++i) { float v = acc[nc][i] + bi; s += v; q += v * v; }
      atomicAdd(&ssum[col], s);
      atomicAdd(&ssq[col], q);
    }
  }
  __syncthreads();
  if (tid < DD) { unsafeAtomicAdd(&gsum[tid], ssum[tid]); unsafeAtomicAdd(&gsq[tid], ssq[tid]); }
}

// pass B: recompute t1, BN+ReLU, m = .@W2+b2, scatter-add into aggr[dst]
__global__ void __launch_bounds__(256) k_edge_passB(const float* __restrict__ h,
                                                    const float* __restrict__ ea,
                                                    const int* __restrict__ srcIdx,
                                                    const int* __restrict__ dstIdx,
                                                    const float* __restrict__ w1,
                                                    const float* __restrict__ b1,
                                                    const float* __restrict__ scl,
                                                    const float* __restrict__ shf,
                                                    const float* __restrict__ w2,
                                                    const float* __restrict__ b2,
                                                    float* __restrict__ aggr) {
  __shared__ _Float16 W1s[DD * 96];
  __shared__ _Float16 W2s[DD * DD];
  __shared__ _Float16 t1s[8][16][DD];        // wave-private 16x64 staging tiles
  __shared__ float b1s[DD], scs[DD], shs[DD], b2s[DD];
  int tid = threadIdx.x;
  for (int idx = tid; idx < DD * 96; idx += 256) {
    int col = idx / 96, k = idx - col * 96;
    W1s[idx] = (k < DD + EFD) ? (_Float16)w1[k * DD + col] : (_Float16)0.f;
  }
  for (int idx = tid; idx < DD * DD; idx += 256) {
    int col = idx >> 6, k = idx & 63;
    W2s[idx] = (_Float16)w2[k * DD + col];
  }
  if (tid < DD) { b1s[tid] = b1[tid]; scs[tid] = scl[tid]; shs[tid] = shf[tid]; b2s[tid] = b2[tid]; }
  __syncthreads();
  int lane = tid & 31, wv = tid >> 5;
  int tile = blockIdx.x * 8 + wv;
  if (tile < NR / 16) {
    v8f acc[4] = {};
    edge_gemm1(h, ea, srcIdx, W1s, tile, lane, acc);
    int cn = lane & 15;
    int base = (lane & 16) ? 8 : 0;
    // BN + ReLU -> wave-private LDS tile (C-layout -> row-major redistribute)
#pragma unroll
    for (int nc = 0; nc < 4; ++nc) {
      int col = nc * 16 + cn;
      float sc = scs[col], sh = shs[col], bi = b1s[col];
#pragma unroll
      for (int i = 0; i < 8; ++i) {
        float v = fmaf(acc[nc][i] + bi, sc, sh);
        t1s[wv][base + i][col] = (_Float16)fmaxf(v, 0.f);
      }
    }
    // LDS is in-order per wave (DScnt); make the RAW explicit.
    asm volatile("s_wait_dscnt 0" ::: "memory");
    // GEMM2: (16x64) @ (64x64)
    const _Float16* trow = &t1s[wv][lane & 15][0];
    v8f acc2[4] = {};
#pragma unroll
    for (int kc = 0; kc < 2; ++kc) {
      v16h a;
      int k0 = kc * 32 + base;
#pragma unroll
      for (int i = 0; i < 8; ++i) { a[i] = trow[k0 + i]; a[8 + i] = trow[k0 + 16 + i]; }
#pragma unroll
      for (int nc = 0; nc < 4; ++nc) {
        v16h b = bfrag(W2s, DD, kc, nc, lane);
        WMMA_F16(acc2[nc], a, b);
      }
    }
    // scatter-add to destination nodes (aggr is L2-resident, 25.6MB)
    int row0 = tile * 16;
    int dsts[8];
#pragma unroll
    for (int i = 0; i < 8; ++i) {
      int r = row0 + base + i;
      dsts[i] = (r < NE) ? dstIdx[r] : (r - NE);
    }
#pragma unroll
    for (int nc = 0; nc < 4; ++nc) {
      int col = nc * 16 + cn;
      float bi = b2s[col];
#pragma unroll
      for (int i = 0; i < 8; ++i)
        unsafeAtomicAdd(&aggr[(size_t)dsts[i] * DD + col], acc2[nc][i] + bi);
    }
  }
}

__global__ void __launch_bounds__(256) k_node_passA(const float* __restrict__ h,
                                                    const float* __restrict__ aggr,
                                                    const float* __restrict__ w1,
                                                    const float* __restrict__ b1,
                                                    float* __restrict__ gsum,
                                                    float* __restrict__ gsq) {
  __shared__ _Float16 W1s[DD * 128];
  __shared__ float bs[DD], ssum[DD], ssq[DD];
  int tid = threadIdx.x;
  for (int idx = tid; idx < DD * 128; idx += 256) {
    int col = idx >> 7, k = idx & 127;
    W1s[idx] = (_Float16)w1[k * DD + col];
  }
  if (tid < DD) { bs[tid] = b1[tid]; ssum[tid] = 0.f; ssq[tid] = 0.f; }
  __syncthreads();
  int lane = tid & 31;
  int tile = blockIdx.x * 8 + (tid >> 5);
  if (tile < NN / 16) {
    v8f acc[4] = {};
    node_gemm1(h, aggr, W1s, tile, lane, acc);
    int cn = lane & 15;
#pragma unroll
    for (int nc = 0; nc < 4; ++nc) {
      int col = nc * 16 + cn;
      float bi = bs[col];
      float s = 0.f, q = 0.f;
#pragma unroll
      for (int i = 0; i < 8; ++i) { float v = acc[nc][i] + bi; s += v; q += v * v; }
      atomicAdd(&ssum[col], s);
      atomicAdd(&ssq[col], q);
    }
  }
  __syncthreads();
  if (tid < DD) { unsafeAtomicAdd(&gsum[tid], ssum[tid]); unsafeAtomicAdd(&gsq[tid], ssq[tid]); }
}

// recompute t2, BN+ReLU, @W2+b2, outer ReLU, write h in place (row-disjoint per wave)
__global__ void __launch_bounds__(256) k_node_passB(float* __restrict__ h,
                                                    const float* __restrict__ aggr,
                                                    const float* __restrict__ w1,
                                                    const float* __restrict__ b1,
                                                    const float* __restrict__ scl,
                                                    const float* __restrict__ shf,
                                                    const float* __restrict__ w2,
                                                    const float* __restrict__ b2) {
  __shared__ _Float16 W1s[DD * 128];
  __shared__ _Float16 W2s[DD * DD];
  __shared__ _Float16 t1s[8][16][DD];
  __shared__ float b1s[DD], scs[DD], shs[DD], b2s[DD];
  int tid = threadIdx.x;
  for (int idx = tid; idx < DD * 128; idx += 256) {
    int col = idx >> 7, k = idx & 127;
    W1s[idx] = (_Float16)w1[k * DD + col];
  }
  for (int idx = tid; idx < DD * DD; idx += 256) {
    int col = idx >> 6, k = idx & 63;
    W2s[idx] = (_Float16)w2[k * DD + col];
  }
  if (tid < DD) { b1s[tid] = b1[tid]; scs[tid] = scl[tid]; shs[tid] = shf[tid]; b2s[tid] = b2[tid]; }
  __syncthreads();
  int lane = tid & 31, wv = tid >> 5;
  int tile = blockIdx.x * 8 + wv;
  if (tile < NN / 16) {
    v8f acc[4] = {};
    node_gemm1(h, aggr, W1s, tile, lane, acc);
    int cn = lane & 15;
    int base = (lane & 16) ? 8 : 0;
#pragma unroll
    for (int nc = 0; nc < 4; ++nc) {
      int col = nc * 16 + cn;
      float sc = scs[col], sh = shs[col], bi = b1s[col];
#pragma unroll
      for (int i = 0; i < 8; ++i) {
        float v = fmaf(acc[nc][i] + bi, sc, sh);
        t1s[wv][base + i][col] = (_Float16)fmaxf(v, 0.f);
      }
    }
    asm volatile("s_wait_dscnt 0" ::: "memory");
    const _Float16* trow = &t1s[wv][lane & 15][0];
    v8f acc2[4] = {};
#pragma unroll
    for (int kc = 0; kc < 2; ++kc) {
      v16h a;
      int k0 = kc * 32 + base;
#pragma unroll
      for (int i = 0; i < 8; ++i) { a[i] = trow[k0 + i]; a[8 + i] = trow[k0 + 16 + i]; }
#pragma unroll
      for (int nc = 0; nc < 4; ++nc) {
        v16h b = bfrag(W2s, DD, kc, nc, lane);
        WMMA_F16(acc2[nc], a, b);
      }
    }
    int row0 = tile * 16;
#pragma unroll
    for (int nc = 0; nc < 4; ++nc) {
      int col = nc * 16 + cn;
      float bi = b2s[col];
#pragma unroll
      for (int i = 0; i < 8; ++i)
        h[(size_t)(row0 + base + i) * DD + col] = fmaxf(acc2[nc][i] + bi, 0.f);
    }
  }
}

__global__ void __launch_bounds__(128) k_bn_finalize(const float* __restrict__ sum,
                                                     const float* __restrict__ sumsq,
                                                     const float* __restrict__ gamma,
                                                     const float* __restrict__ beta,
                                                     float* __restrict__ scl,
                                                     float* __restrict__ shf,
                                                     int C, float invR) {
  int c = threadIdx.x + blockIdx.x * blockDim.x;
  if (c < C) {
    float m = sum[c] * invR;
    float v = sumsq[c] * invR - m * m;
    float sc = gamma[c] * rsqrtf(v + 1e-5f);
    scl[c] = sc;
    shf[c] = beta[c] - m * sc;
  }
}

__global__ void __launch_bounds__(256) k_pool(const float* __restrict__ h, const int* __restrict__ batch,
                                              float* __restrict__ gsum, float* __restrict__ gcnt) {
  long i = (long)blockIdx.x * blockDim.x + threadIdx.x;
  if (i < (long)NN * DD) {
    int n = (int)(i >> 6), c = (int)(i & 63);
    int b = batch[n];
    unsafeAtomicAdd(&gsum[(size_t)b * DD + c], h[i]);
    if (c == 0) unsafeAtomicAdd(&gcnt[b], 1.0f);
  }
}

__global__ void __launch_bounds__(256) k_pool_div(const float* __restrict__ gsum, const float* __restrict__ gcnt,
                                                  float* __restrict__ g) {
  int i = blockIdx.x * blockDim.x + threadIdx.x;
  if (i < NG * DD) g[i] = gsum[i] / fmaxf(gcnt[i >> 6], 1.0f);
}

__global__ void __launch_bounds__(256) k_rgemm(const float* __restrict__ A, const float* __restrict__ W,
                                               const float* __restrict__ bias, float* __restrict__ C,
                                               int rows, int K, int cols) {
  int i = blockIdx.x * blockDim.x + threadIdx.x;
  if (i < rows * cols) {
    int r = i / cols, c = i - r * cols;
    float s = bias[c];
    for (int k = 0; k < K; ++k) s = fmaf(A[(size_t)r * K + k], W[(size_t)k * cols + c], s);
    C[i] = s;
  }
}

__global__ void __launch_bounds__(256) k_colstats(const float* __restrict__ X, int rows, int cols,
                                                  float* __restrict__ sum, float* __restrict__ sumsq) {
  __shared__ float ss[256], sq[256];
  int c = blockIdx.x;
  float s = 0.f, q = 0.f;
  for (int r = threadIdx.x; r < rows; r += 256) {
    float v = X[(size_t)r * cols + c];
    s += v; q += v * v;
  }
  ss[threadIdx.x] = s; sq[threadIdx.x] = q;
  __syncthreads();
  for (int o = 128; o > 0; o >>= 1) {
    if ((int)threadIdx.x < o) { ss[threadIdx.x] += ss[threadIdx.x + o]; sq[threadIdx.x] += sq[threadIdx.x + o]; }
    __syncthreads();
  }
  if (threadIdx.x == 0) { sum[c] = ss[0]; sumsq[c] = sq[0]; }
}

__global__ void __launch_bounds__(256) k_bn_apply_relu(float* __restrict__ X, const float* __restrict__ scl,
                                                       const float* __restrict__ shf, long n, int cols) {
  long i = (long)blockIdx.x * blockDim.x + threadIdx.x;
  if (i < n) {
    int c = (int)(i % cols);
    X[i] = fmaxf(fmaf(X[i], scl[c], shf[c]), 0.f);
  }
}

__global__ void __launch_bounds__(256) k_final(const float* __restrict__ z2, const float* __restrict__ rw3,
                                               const float* __restrict__ rb3, float* __restrict__ out) {
  int g = blockIdx.x * blockDim.x + threadIdx.x;
  if (g < NG) {
    float s = rb3[0];
    for (int k = 0; k < DD; ++k) s = fmaf(z2[(size_t)g * DD + k], rw3[k], s);
    out[g] = s;
  }
}

// ---------------- host launch ----------------

extern "C" void kernel_launch(void* const* d_in, const int* in_sizes, int n_in,
                              void* d_out, int out_size, void* d_ws, size_t ws_size,
                              hipStream_t stream) {
  (void)in_sizes; (void)n_in; (void)out_size; (void)ws_size;
  const float* x     = (const float*)d_in[0];
  const float* ea    = (const float*)d_in[1];
  const float* emb_w = (const float*)d_in[2];
  const float* emb_b = (const float*)d_in[3];
  const float* ew1   = (const float*)d_in[4];
  const float* eb1   = (const float*)d_in[5];
  const float* eg1   = (const float*)d_in[6];
  const float* ebe1  = (const float*)d_in[7];
  const float* ew2   = (const float*)d_in[8];
  const float* eb2   = (const float*)d_in[9];
  const float* uw1   = (const float*)d_in[10];
  const float* ub1   = (const float*)d_in[11];
  const float* ug1   = (const float*)d_in[12];
  const float* ube1  = (const float*)d_in[13];
  const float* uw2   = (const float*)d_in[14];
  const float* ub2   = (const float*)d_in[15];
  const float* rw1   = (const float*)d_in[16];
  const float* rb1   = (const float*)d_in[17];
  const float* rg1   = (const float*)d_in[18];
  const float* rbe1  = (const float*)d_in[19];
  const float* rw2   = (const float*)d_in[20];
  const float* rb2   = (const float*)d_in[21];
  const float* rg2   = (const float*)d_in[22];
  const float* rbe2  = (const float*)d_in[23];
  const float* rw3   = (const float*)d_in[24];
  const float* rb3   = (const float*)d_in[25];
  const int* eidx    = (const int*)d_in[26];
  const int* batch   = (const int*)d_in[27];
  const int* srcI = eidx;
  const int* dstI = eidx + NE;

  // workspace layout (floats), total ~13.5M floats (~54 MB)
  float* ws    = (float*)d_ws;
  float* h     = ws;
  float* aggr  = h + (size_t)NN * DD;
  float* sum   = aggr + (size_t)NN * DD;
  float* sumsq = sum + 128;
  float* scl   = sumsq + 128;
  float* shf   = scl + 128;
  float* gsum  = shf + 128;
  float* gcnt  = gsum + (size_t)NG * DD;
  float* gmean = gcnt + NG;
  float* z1    = gmean + (size_t)NG * DD;
  float* z2    = z1 + (size_t)NG * 128;

  const int edgeBlocks = (NR / 16 + 7) / 8;   // 13282
  const int nodeBlocks = (NN / 16 + 7) / 8;   // 782

  k_embed<<<nodeBlocks, 256, 0, stream>>>(x, emb_w, emb_b, h);

  for (int l = 0; l < 3; ++l) {
    // edge MLP + aggregate
    k_zero<<<1, 256, 0, stream>>>(sum, 256);
    k_edge_passA<<<edgeBlocks, 256, 0, stream>>>(h, ea, srcI, ew1 + (size_t)l * 76 * DD,
                                                 eb1 + l * DD, sum, sumsq);
    k_bn_finalize<<<1, 128, 0, stream>>>(sum, sumsq, eg1 + l * DD, ebe1 + l * DD, scl, shf,
                                         DD, 1.0f / (float)NR);
    k_zero<<<4096, 256, 0, stream>>>(aggr, (long)NN * DD);
    k_edge_passB<<<edgeBlocks, 256, 0, stream>>>(h, ea, srcI, dstI, ew1 + (size_t)l * 76 * DD,
                                                 eb1 + l * DD, scl, shf,
                                                 ew2 + (size_t)l * DD * DD, eb2 + l * DD, aggr);
    // node update
    k_zero<<<1, 256, 0, stream>>>(sum, 256);
    k_node_passA<<<nodeBlocks, 256, 0, stream>>>(h, aggr, uw1 + (size_t)l * 128 * DD,
                                                 ub1 + l * DD, sum, sumsq);
    k_bn_finalize<<<1, 128, 0, stream>>>(sum, sumsq, ug1 + l * DD, ube1 + l * DD, scl, shf,
                                         DD, 1.0f / (float)NN);
    k_node_passB<<<nodeBlocks, 256, 0, stream>>>(h, aggr, uw1 + (size_t)l * 128 * DD,
                                                 ub1 + l * DD, scl, shf,
                                                 uw2 + (size_t)l * DD * DD, ub2 + l * DD);
  }

  // global mean pool
  k_zero<<<1024, 256, 0, stream>>>(gsum, (long)NG * DD + NG);
  k_pool<<<(int)(((long)NN * DD + 255) / 256), 256, 0, stream>>>(h, batch, gsum, gcnt);
  k_pool_div<<<(NG * DD + 255) / 256, 256, 0, stream>>>(gsum, gcnt, gmean);

  // readout MLP (tiny: 2048 rows)
  k_rgemm<<<(NG * 128 + 255) / 256, 256, 0, stream>>>(gmean, rw1, rb1, z1, NG, DD, 128);
  k_colstats<<<128, 256, 0, stream>>>(z1, NG, 128, sum, sumsq);
  k_bn_finalize<<<1, 128, 0, stream>>>(sum, sumsq, rg1, rbe1, scl, shf, 128, 1.0f / (float)NG);
  k_bn_apply_relu<<<(NG * 128 + 255) / 256, 256, 0, stream>>>(z1, scl, shf, (long)NG * 128, 128);

  k_rgemm<<<(NG * DD + 255) / 256, 256, 0, stream>>>(z1, rw2, rb2, z2, NG, 128, DD);
  k_colstats<<<DD, 256, 0, stream>>>(z2, NG, DD, sum, sumsq);
  k_bn_finalize<<<1, 128, 0, stream>>>(sum, sumsq, rg2, rbe2, scl, shf, DD, 1.0f / (float)NG);
  k_bn_apply_relu<<<(NG * DD + 255) / 256, 256, 0, stream>>>(z2, scl, shf, (long)NG * DD, DD);

  k_final<<<(NG + 255) / 256, 256, 0, stream>>>(z2, rw3, rb3, (float*)d_out);
}
